// GCN_81655918231565
// MI455X (gfx1250) — matrix operand order repaired and checked by reading, not compiled, not run on hardware
//
#include <hip/hip_runtime.h>

typedef __attribute__((ext_vector_type(16))) _Float16 v16h;
typedef __attribute__((ext_vector_type(8)))  float    v8f;

#define DH 64

// ---------------- degree / normalization ----------------
__global__ void GCN_deg_kernel(const int* __restrict__ src, const int* __restrict__ dst,
                               float* __restrict__ deg_out, float* __restrict__ deg_in, int E) {
    int e = blockIdx.x * blockDim.x + threadIdx.x;
    if (e < E) {
        atomicAdd(&deg_out[src[e]], 1.0f);
        atomicAdd(&deg_in[dst[e]], 1.0f);
    }
}

__global__ void GCN_invsqrt_kernel(float* __restrict__ a, float* __restrict__ b, int N) {
    int i = blockIdx.x * blockDim.x + threadIdx.x;
    if (i < N) {
        float x = a[i]; a[i] = rsqrtf(x < 1.f ? 1.f : x);
        float y = b[i]; b[i] = rsqrtf(y < 1.f ? 1.f : y);
    }
}

// ---------------- SpMM: gather * inv_sqrt_out[src], scatter-add to dst ----------------
// one wave32 per edge, 2 floats per lane
__global__ void GCN_scatter_kernel(const float* __restrict__ h, const int* __restrict__ src,
                                   const int* __restrict__ dst, const float* __restrict__ inv_out,
                                   float* __restrict__ agg, int E) {
    long t = (long)blockIdx.x * blockDim.x + threadIdx.x;
    int e = (int)(t >> 5);
    int l = (int)(t & 31);
    if (e < E) {
        int s = src[e], d = dst[e];
        float w = inv_out[s];
        const float2 v = *(const float2*)(h + (long)s * DH + l * 2);
        float* outp = agg + (long)d * DH + l * 2;
        atomicAdd(outp,     v.x * w);
        atomicAdd(outp + 1, v.y * w);
    }
}

// ---------------- dense linear via WMMA: relu((agg*inv_in) @ W + b) ----------------
// block = 256 threads = 8 waves; each wave -> 16 nodes x 64 outputs
// per wave: 2 K-steps (K=32) x 4 column tiles = 8 x v_wmma_f32_16x16x32_f16
// W is pre-swizzled into the exact per-lane B-fragment layout in LDS so each
// fragment is one contiguous 32-byte LDS read (2x ds_load_b128).
__global__ void __launch_bounds__(256)
GCN_linear_wmma_kernel(const float* __restrict__ agg, const float* __restrict__ inv_in,
                       const float* __restrict__ W, const float* __restrict__ bias,
                       float* __restrict__ h_out, float* __restrict__ xs_out,
                       int N, int layer) {
    // Wsw[tile][lane][h]: tile = kk*4 + c; lane's B fragment for that 16x16x32 tile.
    // B fragment as built here (k order matches A): lane l -> n = l&15, K = (l>>4)*16 + h (+ kk*32).
    __shared__ _Float16 Wsw[8 * 32 * 16];
    __shared__ float    bs[DH];

    {
        // 256 threads, 256 (tile,lane) slots: each thread builds one fragment (16 halves).
        const int i  = threadIdx.x;      // i = tile*32 + lane
        const int tl = i >> 5;           // tile 0..7
        const int ll = i & 31;           // lane 0..31
        const int kk = tl >> 2;
        const int c  = tl & 3;
        const int kbase = kk * 32 + (ll >> 4) * 16;
        const int ncol  = c * 16 + (ll & 15);
        _Float16* dstp = &Wsw[i * 16];
#pragma unroll
        for (int h = 0; h < 16; ++h)
            dstp[h] = (_Float16)W[(kbase + h) * DH + ncol];
    }
    if (threadIdx.x < DH) bs[threadIdx.x] = bias[threadIdx.x];
    __syncthreads();

    const int lane = threadIdx.x & 31;
    const int wave = threadIdx.x >> 5;
    const int g    = lane >> 4;    // half-wave group
    const int mn   = lane & 15;    // M index for A, N index for B/C
    const int node_base = blockIdx.x * 128 + wave * 16;

    // A-row pointer: clamp OOB node to row 0, zero its scale -> contributes 0
    const int node_a  = node_base + mn;
    const int na      = (node_a < N) ? node_a : 0;
    float sc          = inv_in[na];
    if (node_a >= N) sc = 0.f;
    const float* arow = agg + (long)na * DH;

    v8f acc[4] = {};   // 4 column tiles of 16x16 f32

#pragma unroll
    for (int kk = 0; kk < 2; ++kk) {
        // A fragment: lane holds A[m][kk*32 + g*16 + h], h = 0..15
        v16h a;
        const int k0 = kk * 32 + g * 16;
#pragma unroll
        for (int h = 0; h < 8; ++h) {
            a[h]     = (_Float16)(arow[k0 + h]      * sc);
            a[h + 8] = (_Float16)(arow[k0 + 8 + h] * sc);
        }
#pragma unroll
        for (int c = 0; c < 4; ++c) {
            // whole B fragment contiguous in LDS: 32 bytes per lane
            const v16h b = *(const v16h*)(&Wsw[((kk * 4 + c) * 32 + lane) * 16]);
            acc[c] = __builtin_amdgcn_wmma_f32_16x16x32_f16(
                false, a, false, b, (short)0, acc[c], false, false);
        }
    }

    // Epilogue: VGPR r of acc[c] holds C[M = r + g*8][Nc = mn]
    const int node_c = node_base + g * 8;
#pragma unroll
    for (int r = 0; r < 8; ++r) {
        const int node = node_c + r;
        if (node < N) {
#pragma unroll
            for (int c = 0; c < 4; ++c) {
                const int f = c * 16 + mn;
                float v = acc[c][r] + bs[f];
                v = v > 0.f ? v : 0.f;
                h_out[(long)node * DH + f] = v;                       // next-layer input
                xs_out[(long)node * (3 * DH) + layer * DH + f] = v;   // xs_cat slice
            }
        }
    }
}

// ---------------- per-node row sum (x_pool) ----------------
__global__ void GCN_pool_kernel(const float* __restrict__ h, float* __restrict__ out, int N) {
    int wave = threadIdx.x >> 5;
    int lane = threadIdx.x & 31;
    int n = blockIdx.x * (blockDim.x >> 5) + wave;
    if (n < N) {
        const float* row = h + (long)n * DH;
        float s = row[lane] + row[lane + 32];
#pragma unroll
        for (int off = 16; off > 0; off >>= 1) s += __shfl_down(s, off, 32);
        if (lane == 0) out[n] = s;
    }
}

extern "C" void kernel_launch(void* const* d_in, const int* in_sizes, int n_in,
                              void* d_out, int out_size, void* d_ws, size_t ws_size,
                              hipStream_t stream) {
    const float* node_feat = (const float*)d_in[0];
    const int*   src       = (const int*)d_in[1];
    const int*   dst       = (const int*)d_in[2];
    // d_in[3] = graph_len (unused; N derived from node_feat size)
    const float* Wp[3] = {(const float*)d_in[4], (const float*)d_in[6], (const float*)d_in[8]};
    const float* bp[3] = {(const float*)d_in[5], (const float*)d_in[7], (const float*)d_in[9]};

    const int N = in_sizes[0] / DH;
    const int E = in_sizes[1];

    // workspace layout
    char* ws = (char*)d_ws;
    float* inv_out = (float*)ws;  ws += (size_t)N * sizeof(float);
    float* inv_in  = (float*)ws;  ws += (size_t)N * sizeof(float);
    float* agg     = (float*)ws;  ws += (size_t)N * DH * sizeof(float);
    float* hA      = (float*)ws;  ws += (size_t)N * DH * sizeof(float);
    float* hB      = (float*)ws;

    float* out    = (float*)d_out;
    float* xs_out = out + (size_t)3 * N;   // xs_cat region [N, 192]

    // degrees -> inv-sqrt norms (inv_out/inv_in contiguous: single memset)
    hipMemsetAsync(inv_out, 0, (size_t)2 * N * sizeof(float), stream);
    GCN_deg_kernel<<<(E + 255) / 256, 256, 0, stream>>>(src, dst, inv_out, inv_in, E);
    GCN_invsqrt_kernel<<<(N + 255) / 256, 256, 0, stream>>>(inv_out, inv_in, N);

    const float* h_in = node_feat;
    float* bufs[2] = {hA, hB};
    for (int l = 0; l < 3; ++l) {
        hipMemsetAsync(agg, 0, (size_t)N * DH * sizeof(float), stream);

        unsigned long sthreads = (unsigned long)E * 32ul;
        GCN_scatter_kernel<<<(unsigned)((sthreads + 255) / 256), 256, 0, stream>>>(
            h_in, src, dst, inv_out, agg, E);

        float* h_next = bufs[l & 1];
        GCN_linear_wmma_kernel<<<(N + 127) / 128, 256, 0, stream>>>(
            agg, inv_in, Wp[l], bp[l], h_next, xs_out, N, l);

        GCN_pool_kernel<<<(N + 7) / 8, 256, 0, stream>>>(h_next, out + (size_t)l * N, N);

        h_in = h_next;
    }
}
